// PV_LSTM_22445499089678
// MI455X (gfx1250) — matrix operand order repaired and checked by reading, not compiled, not run on hardware
//
#include <hip/hip_runtime.h>
#include <cstdint>

#define DEVFN static __device__ __forceinline__

typedef __bf16 bf16_t;
typedef bf16_t v16bf __attribute__((ext_vector_type(16)));
typedef float  v8f   __attribute__((ext_vector_type(8)));

union Frag { uint4 q[2]; v16bf v; };

constexpr int Hn    = 256;    // hidden size
constexpr int TOBS  = 16;     // observed steps
constexpr int TOUT  = 16;     // output steps
constexpr int Bn    = 16384;  // batch
constexpr int HS    = 264;    // h row stride in bf16 elems (bank rotate 4/row, 16B aligned)
constexpr int GATES = 4 * Hn; // 1024
constexpr int MT    = 4;      // M-tiles of 16 rows -> 64 batch rows per WG
constexpr int WGR   = 16 * MT;
constexpr int NW    = 8;      // waves per WG (2 waves/SIMD -> ~512 VGPR budget)
constexpr int NTHR  = NW * 32;
constexpr int HB_ELE = WGR * HS;   // 16896 ushorts per h buffer
constexpr int XB_ELE = WGR * 32;   // 2048 ushorts per x buffer (32 cols; col4 = 1.0)

DEVFN unsigned short f2bf(float f) {
  unsigned int u = __float_as_uint(f);
  u += 0x7FFFu + ((u >> 16) & 1u);            // round-to-nearest-even
  return (unsigned short)(u >> 16);
}
DEVFN float bf2f(unsigned short h) { return __uint_as_float(((unsigned int)h) << 16); }

DEVFN float rcpf(float x) { return __builtin_amdgcn_rcpf(x); }          // v_rcp_f32
DEVFN float sigm(float x) { return rcpf(1.0f + __expf(-x)); }
DEVFN float tanh_f(float x) {
  float t = __expf(-2.0f * fabsf(x));
  float r = (1.0f - t) * rcpf(1.0f + t);
  return copysignf(r, x);
}

DEVFN v8f wmma_bf16(v16bf a, v16bf b, v8f c) {
  return __builtin_amdgcn_wmma_f32_16x16x32_bf16(false, a, false, b, (short)0, c, false, false);
}

// A fragment (16x32 bf16) from LDS row-major buffer (stride in elems).
// lanes 0-15 -> M=lane, K base 0; lanes 16-31 -> M=lane-16, K base 8;
// VGPR0-3 = K[base..base+7], VGPR4-7 = K[base+16..base+23].  (two ds_load_b128)
DEVFN v16bf load_a(const unsigned short* buf, int stride, int k0, int laneLo, int laneHi) {
  const unsigned short* row = buf + laneLo * stride + k0 + laneHi * 8;
  Frag t;
  t.q[0] = *(const uint4*)row;
  t.q[1] = *(const uint4*)(row + 16);
  return t.v;
}

// B fragment (32x16 bf16) for g = h @ W.T : B[k][n] = W[nBase+n][k0+k].
// lanes 0-15 -> N=lane, K=0..7,16..23; lanes 16-31 -> N=lane-16, K=8..15,24..31.
DEVFN v16bf load_b_glb(const unsigned short* __restrict__ W, int nBase, int rowStride, int k0,
                       int laneLo, int laneHi) {
  const unsigned short* row = W + (size_t)(nBase + laneLo) * rowStride + k0 + laneHi * 8;
  Frag t;
  t.q[0] = *(const uint4*)row;
  t.q[1] = *(const uint4*)(row + 16);
  return t.v;
}

// GEMM for one gate: acc[hc][mt]. Wave w owns hidden [w*32, w*32+32).
// Bias comes in through the x-chunk (x col4 == 1.0, wihp col4 == fused bias),
// so the first WMMA per tile starts from a literal-zero C.
DEVFN void gemm_gate(const unsigned short* __restrict__ whh,
                     const unsigned short* __restrict__ wihp,
                     const unsigned short* hbuf, const unsigned short* xbuf,
                     int g, int w, int laneLo, int laneHi,
                     v8f acc[2][MT]) {
  const v8f zero = {0.f, 0.f, 0.f, 0.f, 0.f, 0.f, 0.f, 0.f};
  {  // x-term (K=32 padded chunk, includes bias)
    v16bf bx[2];
#pragma unroll
    for (int hc = 0; hc < 2; ++hc)
      bx[hc] = load_b_glb(wihp, g * Hn + w * 32 + hc * 16, 32, 0, laneLo, laneHi);
#pragma unroll
    for (int mt = 0; mt < MT; ++mt) {
      v16bf a = load_a(xbuf + mt * 16 * 32, 32, 0, laneLo, laneHi);
#pragma unroll
      for (int hc = 0; hc < 2; ++hc) acc[hc][mt] = wmma_bf16(a, bx[hc], zero);
    }
  }
#pragma unroll
  for (int kc = 0; kc < 8; ++kc) {  // h-term: B loaded once, reused by 4 M-tiles
    v16bf bh[2];
#pragma unroll
    for (int hc = 0; hc < 2; ++hc)
      bh[hc] = load_b_glb(whh, g * Hn + w * 32 + hc * 16, Hn, kc * 32, laneLo, laneHi);
#pragma unroll
    for (int mt = 0; mt < MT; ++mt) {
      v16bf a = load_a(hbuf + mt * 16 * HS, HS, kc * 32, laneLo, laneHi);
#pragma unroll
      for (int hc = 0; hc < 2; ++hc) acc[hc][mt] = wmma_bf16(a, bh[hc], acc[hc][mt]);
    }
  }
}

// One LSTM step, gate-serial f -> i -> g -> o; writes new h into hnew (ping-pong).
DEVFN void lstm_step(const unsigned short* __restrict__ whh,
                     const unsigned short* __restrict__ wihp,
                     const unsigned short* hbuf, const unsigned short* xbuf,
                     unsigned short* hnew,
                     float c[2][MT][8], int w, int laneLo, int laneHi) {
  v8f acc[2][MT];
  gemm_gate(whh, wihp, hbuf, xbuf, 1, w, laneLo, laneHi, acc);   // f
#pragma unroll
  for (int hc = 0; hc < 2; ++hc)
#pragma unroll
    for (int mt = 0; mt < MT; ++mt)
#pragma unroll
      for (int e = 0; e < 8; ++e) c[hc][mt][e] *= sigm(acc[hc][mt][e]);

  float si[2][MT][8];
  gemm_gate(whh, wihp, hbuf, xbuf, 0, w, laneLo, laneHi, acc);   // i
#pragma unroll
  for (int hc = 0; hc < 2; ++hc)
#pragma unroll
    for (int mt = 0; mt < MT; ++mt)
#pragma unroll
      for (int e = 0; e < 8; ++e) si[hc][mt][e] = sigm(acc[hc][mt][e]);

  gemm_gate(whh, wihp, hbuf, xbuf, 2, w, laneLo, laneHi, acc);   // g
#pragma unroll
  for (int hc = 0; hc < 2; ++hc)
#pragma unroll
    for (int mt = 0; mt < MT; ++mt)
#pragma unroll
      for (int e = 0; e < 8; ++e) c[hc][mt][e] += si[hc][mt][e] * tanh_f(acc[hc][mt][e]);

  gemm_gate(whh, wihp, hbuf, xbuf, 3, w, laneLo, laneHi, acc);   // o
#pragma unroll
  for (int hc = 0; hc < 2; ++hc) {
    int j = w * 32 + hc * 16 + laneLo;
#pragma unroll
    for (int mt = 0; mt < MT; ++mt)
#pragma unroll
      for (int e = 0; e < 8; ++e) {
        float h = sigm(acc[hc][mt][e]) * tanh_f(c[hc][mt][e]);
        hnew[(mt * 16 + e + 8 * laneHi) * HS + j] = f2bf(h);
      }
  }
}

__global__ __launch_bounds__(NTHR)
void pv_lstm_main(const float* __restrict__ speed, const float* __restrict__ pos,
                  const unsigned short* __restrict__ whh4,
                  const unsigned short* __restrict__ wihp4,
                  const unsigned short* __restrict__ head_sp,
                  const unsigned short* __restrict__ head_cr,
                  const float* __restrict__ hb_sp, const float* __restrict__ hb_cr,
                  unsigned short* __restrict__ bufH,   // [Bn][Hn] bf16 phase-bridge (h)
                  unsigned short* __restrict__ bufC,   // [Bn][Hn] bf16 phase-bridge (c)
                  float* __restrict__ out) {
  extern __shared__ unsigned short smem[];
  // layout: [hbuf0 | hbuf1 | xbuf0 | xbuf1]  (pointers derived per use; no pointer arrays)

  const int tid    = threadIdx.x;
  const int w      = tid >> 5;
  const int lane   = tid & 31;
  const int laneLo = lane & 15;
  const int laneHi = lane >> 4;
  const int rowBase = blockIdx.x * WGR;

  float c[2][MT][8];

  // x buffers: zero, with constant 1.0 in col 4 (bias input column)
  for (int i = tid; i < 2 * XB_ELE; i += NTHR)
    smem[2 * HB_ELE + i] = ((i & 31) == 4) ? (unsigned short)0x3F80 : (unsigned short)0;

  // Phases: 0=enc-speed, 1=enc-pos, 2=dec-speed, 3=dec-cross
  for (int p = 0; p < 4; ++p) {
    __syncthreads();
    const unsigned short* whh  = whh4  + (size_t)p * ((size_t)GATES * Hn);
    const unsigned short* wihp = wihp4 + (size_t)p * ((size_t)GATES * 32);
    const float* seq  = (p == 0 || p == 2) ? speed : pos;
    unsigned short* h0buf = smem;                 // ping-pong slot 0
    unsigned short* x0buf = smem + 2 * HB_ELE;    // x slot 0

    if (p < 2) {
      for (int i = tid; i < HB_ELE; i += NTHR) h0buf[i] = 0;
#pragma unroll
      for (int hc = 0; hc < 2; ++hc)
#pragma unroll
        for (int mt = 0; mt < MT; ++mt)
#pragma unroll
          for (int e = 0; e < 8; ++e) c[hc][mt][e] = 0.f;
    } else {
      if (p == 3) {  // restore h0/c0 from bridge
#pragma unroll
        for (int hc = 0; hc < 2; ++hc) {
          int j = w * 32 + hc * 16 + laneLo;
#pragma unroll
          for (int mt = 0; mt < MT; ++mt)
#pragma unroll
            for (int e = 0; e < 8; ++e) {
              size_t gidx = (size_t)(rowBase + mt * 16 + e + 8 * laneHi) * Hn + j;
              c[hc][mt][e] = bf2f(bufC[gidx]);
              h0buf[(mt * 16 + e + 8 * laneHi) * HS + j] = bufH[gidx];
            }
        }
      }
      // (p==2: h slot0 / c already hold h0/c0 from the p==1 epilogue)
      if (tid < WGR * 4) {  // decoder x init = last observed input
        int r = tid >> 2, s = tid & 3;
        x0buf[r * 32 + s] = f2bf(seq[((size_t)(rowBase + r) * TOBS + (TOBS - 1)) * 4 + s]);
      }
    }

    int par = 0;
    for (int t = 0; t < TOUT; ++t) {
      unsigned short* hcur = smem + par * HB_ELE;
      unsigned short* hnxt = smem + (par ^ 1) * HB_ELE;
      unsigned short* xcur = smem + 2 * HB_ELE + par * XB_ELE;
      unsigned short* xnxt = smem + 2 * HB_ELE + (par ^ 1) * XB_ELE;

      if (p < 2 && tid < WGR * 4) {  // encoder input for step t
        int r = tid >> 2, s = tid & 3;
        xcur[r * 32 + s] = f2bf(seq[((size_t)(rowBase + r) * TOBS + t) * 4 + s]);
      }
      __syncthreads();  // previous step's reads done; this step's h/x visible
      lstm_step(whh, wihp, hcur, xcur, hnxt, c, w, laneLo, laneHi);
      if (p >= 2) {
        __syncthreads();  // new h (hnxt) visible for head
        if (w < MT) {     // waves 0..3: one head M-tile each
          const int mt = w;
          const unsigned short* HW = (p == 2) ? head_sp : head_cr;
          float hbias = ((p == 2) ? hb_sp : hb_cr)[laneLo];
          v8f acc;
#pragma unroll
          for (int e = 0; e < 8; ++e) acc[e] = hbias;
#pragma unroll
          for (int kc = 0; kc < 8; ++kc) {
            v16bf a = load_a(hnxt + mt * 16 * HS, HS, kc * 32, laneLo, laneHi);
            acc = wmma_bf16(a, load_b_glb(HW, 0, Hn, kc * 32, laneLo, laneHi), acc);
          }
#pragma unroll
          for (int r = 0; r < 8; ++r) {
            int m = mt * 16 + r + 8 * laneHi;
            size_t b = (size_t)(rowBase + m);
            if (p == 2) {                       // speed: hardtanh, fed back as x
              if (laneLo < 4) {
                float vv = fminf(fmaxf(acc[r], -100.0f), 100.0f);
                out[(b * TOUT + t) * 4 + laneLo] = vv;
                xnxt[m * 32 + laneLo] = f2bf(vv);
              }
            } else {                            // crossing: softmax(relu); x = relu(emb)
              float vr = fmaxf(acc[r], 0.0f);
              float pr = __shfl_xor(vr, 1, 32);
              if (laneLo < 2) {
                float mx = fmaxf(vr, pr);
                float e0 = __expf(vr - mx), e1 = __expf(pr - mx);
                out[(size_t)Bn * TOUT * 4 + (b * TOUT + t) * 2 + laneLo] = e0 * rcpf(e0 + e1);
              }
              if (laneLo >= 4 && laneLo < 8)
                xnxt[m * 32 + (laneLo - 4)] = f2bf(vr);
            }
          }
        }
      }
      par ^= 1;
    }
    // par is back to 0 here; final h of this phase lives in slot 0.

    // ---- phase epilogues: bridge state through global ws (one-shot, tiny traffic)
    if (p == 0) {        // save speed-encoder finals (own slice; self-written)
#pragma unroll
      for (int hc = 0; hc < 2; ++hc) {
        int j = w * 32 + hc * 16 + laneLo;
#pragma unroll
        for (int mt = 0; mt < MT; ++mt)
#pragma unroll
          for (int e = 0; e < 8; ++e) {
            size_t gidx = (size_t)(rowBase + mt * 16 + e + 8 * laneHi) * Hn + j;
            bufH[gidx] = h0buf[(mt * 16 + e + 8 * laneHi) * HS + j];
            bufC[gidx] = f2bf(c[hc][mt][e]);
          }
      }
    } else if (p == 1) { // h0 = h_sp + h_po ; c0 = c_sp + c_po ; stash + stage for p2
#pragma unroll
      for (int hc = 0; hc < 2; ++hc) {
        int j = w * 32 + hc * 16 + laneLo;
#pragma unroll
        for (int mt = 0; mt < MT; ++mt)
#pragma unroll
          for (int e = 0; e < 8; ++e) {
            int lidx = (mt * 16 + e + 8 * laneHi) * HS + j;
            size_t gidx = (size_t)(rowBase + mt * 16 + e + 8 * laneHi) * Hn + j;
            float h0 = bf2f(bufH[gidx]) + bf2f(h0buf[lidx]);
            float c0 = bf2f(bufC[gidx]) + c[hc][mt][e];
            unsigned short h0b = f2bf(h0);
            bufH[gidx] = h0b;
            bufC[gidx] = f2bf(c0);
            h0buf[lidx] = h0b;     // stage h0 for p2 (own slice)
            c[hc][mt][e] = c0;
          }
      }
    }
  }
}

// ---------------- weight prep kernels ----------------
__global__ void k_cvt_bf16(const float* __restrict__ src, unsigned short* __restrict__ dst, int n) {
  int i = blockIdx.x * blockDim.x + threadIdx.x;
  if (i < n) dst[i] = f2bf(src[i]);
}
// Wih padded to 32 cols; col 4 carries the fused bias (bih+bhh).
__global__ void k_pad_wih(const float* __restrict__ src, const float* __restrict__ bih,
                          const float* __restrict__ bhh, unsigned short* __restrict__ dst) {
  int i = blockIdx.x * blockDim.x + threadIdx.x;  // over 1024*32
  if (i < GATES * 32) {
    int r = i >> 5, col = i & 31;
    unsigned short v = 0;
    if (col < 4)       v = f2bf(src[r * 4 + col]);
    else if (col == 4) v = f2bf(bih[r] + bhh[r]);
    dst[i] = v;
  }
}
__global__ void k_heads(const float* __restrict__ fsW, const float* __restrict__ fsb,
                        const float* __restrict__ fcW, const float* __restrict__ fcb,
                        const float* __restrict__ emW, const float* __restrict__ emb,
                        unsigned short* __restrict__ head_sp, unsigned short* __restrict__ head_cr,
                        float* __restrict__ hb_sp, float* __restrict__ hb_cr) {
  int i = blockIdx.x * blockDim.x + threadIdx.x;
  if (i < 16 * Hn) {
    int r = i >> 8, col = i & 255;
    head_sp[i] = (r < 4) ? f2bf(fsW[r * Hn + col]) : (unsigned short)0;
    unsigned short v = 0;
    if (r < 2)                v = f2bf(fcW[r * Hn + col]);
    else if (r >= 4 && r < 8) v = f2bf(emW[(r - 4) * Hn + col]);
    head_cr[i] = v;
  }
  if (i < 16) {
    hb_sp[i] = (i < 4) ? fsb[i] : 0.f;
    float v = 0.f;
    if (i < 2)                v = fcb[i];
    else if (i >= 4 && i < 8) v = emb[i - 4];
    hb_cr[i] = v;
  }
}

extern "C" void kernel_launch(void* const* d_in, const int* in_sizes, int n_in,
                              void* d_out, int out_size, void* d_ws, size_t ws_size,
                              hipStream_t stream) {
  (void)in_sizes; (void)n_in; (void)out_size;
  char* ws = (char*)d_ws;
  size_t off = 0;
  auto take = [&](size_t bytes) {
    char* p = ws + off;
    off = (off + bytes + 255) & ~(size_t)255;
    return p;
  };

  unsigned short* whh4    = (unsigned short*)take(4 * (size_t)GATES * Hn * 2);  // 2 MB
  unsigned short* wihp4   = (unsigned short*)take(4 * (size_t)GATES * 32 * 2);  // 256 KB
  unsigned short* head_sp = (unsigned short*)take(16 * Hn * 2);
  unsigned short* head_cr = (unsigned short*)take(16 * Hn * 2);
  float*          hb_sp   = (float*)take(16 * 4);
  float*          hb_cr   = (float*)take(16 * 4);
  unsigned short* bufH    = (unsigned short*)take((size_t)Bn * Hn * 2);         // 8 MB
  unsigned short* bufC    = (unsigned short*)take((size_t)Bn * Hn * 2);         // 8 MB
  if (off > ws_size) return;  // workspace too small; nothing safe to do

  // input order: sp(2..5), po(6..9), dsp(10..13), dcr(14..17)
  const int whh_idx[4] = {3, 7, 11, 15};
  const int wih_idx[4] = {2, 6, 10, 14};
  const int bih_idx[4] = {4, 8, 12, 16};

  for (int m = 0; m < 4; ++m) {
    k_cvt_bf16<<<(GATES * Hn + 255) / 256, 256, 0, stream>>>(
        (const float*)d_in[whh_idx[m]], whh4 + (size_t)m * GATES * Hn, GATES * Hn);
    k_pad_wih<<<(GATES * 32 + 255) / 256, 256, 0, stream>>>(
        (const float*)d_in[wih_idx[m]], (const float*)d_in[bih_idx[m]],
        (const float*)d_in[bih_idx[m] + 1], wihp4 + (size_t)m * GATES * 32);
  }
  k_heads<<<(16 * Hn + 255) / 256, 256, 0, stream>>>(
      (const float*)d_in[18], (const float*)d_in[19], (const float*)d_in[20],
      (const float*)d_in[21], (const float*)d_in[22], (const float*)d_in[23],
      head_sp, head_cr, hb_sp, hb_cr);

  size_t ldsBytes = (size_t)(2 * HB_ELE + 2 * XB_ELE) * sizeof(unsigned short);  // ~74 KB
  pv_lstm_main<<<Bn / WGR, NTHR, ldsBytes, stream>>>(
      (const float*)d_in[0], (const float*)d_in[1],
      whh4, wihp4, head_sp, head_cr, hb_sp, hb_cr, bufH, bufC, (float*)d_out);
}
